// Block_40475771797636
// MI455X (gfx1250) — matrix operand order repaired and checked by reading, not compiled, not run on hardware
//
#include <hip/hip_runtime.h>
#include <hip/hip_bf16.h>
#include <math.h>
#include <stdint.h>

typedef __attribute__((ext_vector_type(16))) __bf16 v16bf;
typedef __attribute__((ext_vector_type(8)))  __bf16 v8bf;
typedef __attribute__((ext_vector_type(8)))  float  v8f;

namespace {
constexpr int kB   = 2;
constexpr int kN   = 1569;   // 1 + 8*196
constexpr int kC   = 768;
constexpr int kH   = 12;
constexpr int kD   = 64;
constexpr int kF   = 8;
constexpr int kP   = 196;
constexpr int kS   = 1568;   // F*P
constexpr int kHid = 3072;
constexpr int kBH  = kB * kH;        // 24
constexpr int kM1  = kB * kN;        // 3138
constexpr int kMS  = kB * kS;        // 3136
constexpr int kMX  = kB * kS * kF;   // 25088
constexpr int kVTROW = kF * 224;     // 1792 (per-frame padded to 224)
constexpr float kScale = 0.125f;     // d^-0.5, d=64
}

union AFrag { v16bf v; v8bf h[2]; };

static __device__ __forceinline__ v8f wmma_bf16(v16bf a, v16bf b, v8f c) {
  return __builtin_amdgcn_wmma_f32_16x16x32_bf16(false, a, false, b, (short)0, c,
                                                 false, false);
}

// async copy 32 bytes global -> LDS (two B128 transfers), tracked by ASYNCcnt
static __device__ __forceinline__ void async_copy32(uint32_t lds_off,
                                                    const __bf16* gptr) {
  asm volatile(
      "global_load_async_to_lds_b128 %0, %1, off\n\t"
      "global_load_async_to_lds_b128 %0, %1, off offset:16"
      :: "v"(lds_off), "v"(gptr) : "memory");
}
static __device__ __forceinline__ void wait_async0() {
  asm volatile("s_wait_asynccnt 0x0" ::: "memory");
}

// ---------------- weight transpose + bf16 convert: dst[n*K + k] = src[k*ld + col0 + n]
__global__ __launch_bounds__(256) void transpose_w_kernel(
    const float* __restrict__ src, __bf16* __restrict__ dst,
    int K, int N, int ld, int col0) {
  long idx = (long)blockIdx.x * 256 + threadIdx.x;
  if (idx >= (long)N * K) return;
  int n = (int)(idx / K);
  int k = (int)(idx % K);
  dst[idx] = (__bf16)src[(size_t)k * ld + col0 + n];
}

// ---------------- LayerNorm (row = block), C=768 = 256*3, output bf16
__global__ __launch_bounds__(256) void ln_kernel(
    const float* __restrict__ x, const float* __restrict__ g,
    const float* __restrict__ beta, __bf16* __restrict__ out) {
  int row = blockIdx.x;
  int t = threadIdx.x;
  const float* xr = x + (size_t)row * kC;
  __shared__ float red[256];
  float v0 = xr[t], v1 = xr[t + 256], v2 = xr[t + 512];
  red[t] = v0 + v1 + v2;
  __syncthreads();
  for (int o = 128; o > 0; o >>= 1) { if (t < o) red[t] += red[t + o]; __syncthreads(); }
  float mu = red[0] * (1.0f / 768.0f);
  __syncthreads();
  float d0 = v0 - mu, d1 = v1 - mu, d2 = v2 - mu;
  red[t] = d0 * d0 + d1 * d1 + d2 * d2;
  __syncthreads();
  for (int o = 128; o > 0; o >>= 1) { if (t < o) red[t] += red[t + o]; __syncthreads(); }
  float rinv = rsqrtf(red[0] * (1.0f / 768.0f) + 1e-5f);
  __bf16* orow = out + (size_t)row * kC;
  orow[t]       = (__bf16)(d0 * rinv * g[t]       + beta[t]);
  orow[t + 256] = (__bf16)(d1 * rinv * g[t + 256] + beta[t + 256]);
  orow[t + 512] = (__bf16)(d2 * rinv * g[t + 512] + beta[t + 512]);
}

// ---------------- generic bf16 WMMA GEMM: out = A(MxK) * Bt(NxK)^T with epilogue
// block tile 128x64, 8 waves, each wave 32x32 (2x2 wmma 16x16 tiles), K step 32
// Tile staging uses GLOBAL_LOAD_ASYNC_TO_LDS_B128 + s_wait_asynccnt (CDNA5 path).
__global__ __launch_bounds__(256) void gemm_bf16_kernel(
    const __bf16* __restrict__ A, const __bf16* __restrict__ Bt,
    int M, int N, int K,
    const float* __restrict__ bias, const float* __restrict__ resid,
    float* __restrict__ outF, __bf16* __restrict__ outH, int act_gelu) {
  __shared__ __bf16 As[128][48];   // pitch 48 bf16 = 96B (16B-aligned everywhere)
  __shared__ __bf16 Bs[64][48];
  const int t = threadIdx.x;
  const int lane = t & 31;
  const int w = t >> 5;
  const int wr = w & 3;            // wave row tile (4)
  const int wc = w >> 2;           // wave col tile (2)
  const int lr = lane & 15;
  const int lh = lane >> 4;
  const int m0 = blockIdx.x * 128;
  const int n0 = blockIdx.y * 64;

  v8f acc[2][2];
#pragma unroll
  for (int r = 0; r < 2; ++r)
#pragma unroll
    for (int c = 0; c < 2; ++c) acc[r][c] = {};

  const int ldrow = t >> 1;
  const int ldhalf = (t & 1) * 16;
  // OOB A rows clamp to last valid row: they only feed discarded C rows.
  int gr = m0 + ldrow;
  if (gr >= M) gr = M - 1;
  const __bf16* gA = A + (size_t)gr * K + ldhalf;
  const __bf16* gB = Bt + (size_t)(n0 + ldrow) * K + ldhalf;
  const uint32_t ldsA = (uint32_t)(uintptr_t)&As[ldrow][ldhalf];
  const uint32_t ldsB = (uint32_t)(uintptr_t)&Bs[ldrow][ldhalf];
  const bool doB = (t < 128);      // wave-uniform (waves 0..3)

  for (int kb = 0; kb < K; kb += 32) {
    __syncthreads();               // previous-iter readers done before overwrite
    async_copy32(ldsA, gA + kb);
    if (doB) async_copy32(ldsB, gB + kb);
    wait_async0();
    __syncthreads();               // all waves' tiles landed in LDS
    AFrag a[2];
#pragma unroll
    for (int r = 0; r < 2; ++r) {
      const __bf16* ap = &As[wr * 32 + r * 16 + lr][lh * 16];
      a[r].h[0] = *(const v8bf*)(ap);
      a[r].h[1] = *(const v8bf*)(ap + 8);
    }
    v16bf bfr[2];
#pragma unroll
    for (int c = 0; c < 2; ++c)
      bfr[c] = *(const v16bf*)(&Bs[wc * 32 + c * 16 + lr][lh * 16]);
#pragma unroll
    for (int r = 0; r < 2; ++r)
#pragma unroll
      for (int c = 0; c < 2; ++c)
        acc[r][c] = wmma_bf16(a[r].v, bfr[c], acc[r][c]);
  }

#pragma unroll
  for (int r = 0; r < 2; ++r) {
#pragma unroll
    for (int c = 0; c < 2; ++c) {
#pragma unroll
      for (int i = 0; i < 8; ++i) {
        int gm = m0 + wr * 32 + r * 16 + lh * 8 + i;
        int gn = n0 + wc * 32 + c * 16 + lr;
        if (gm < M) {
          float v = acc[r][c][i];
          if (bias)     v += bias[gn];
          if (act_gelu) v = 0.5f * v * (1.0f + erff(v * 0.70710678118f));
          if (resid)    v += resid[(size_t)gm * N + gn];
          if (outF)     outF[(size_t)gm * N + gn] = v;
          if (outH)     outH[(size_t)gm * N + gn] = (__bf16)v;
        }
      }
    }
  }
}

// ---------------- gather q/k (bf16, layout (bh, s, d)); zero 32-row tail pad of k
__global__ __launch_bounds__(256) void gather_qk_kernel(
    const float* __restrict__ qkv, __bf16* __restrict__ q, __bf16* __restrict__ k) {
  int idx = blockIdx.x * 256 + threadIdx.x;
  const int rows = kBH * kS + 32;
  if (idx >= rows * kD) return;
  int row = idx >> 6;
  if (row >= kBH * kS) { k[idx] = (__bf16)0.0f; return; }
  int s = row % kS;
  int bh = row / kS;
  int b = bh / kH, h = bh % kH;
  size_t base = ((size_t)(b * kN + 1 + s)) * (3 * kC) + h * kD + (idx & 63);
  q[idx] = (__bf16)qkv[base];
  k[idx] = (__bf16)qkv[base + kC];
}

// ---------------- gather v transposed (bh, d, F*224) with zero padding per frame
__global__ __launch_bounds__(256) void gather_v_kernel(
    const float* __restrict__ qkv, __bf16* __restrict__ vT) {
  long idx = (long)blockIdx.x * 256 + threadIdx.x;
  if (idx >= (long)kBH * kD * kVTROW) return;
  int j  = (int)(idx % kVTROW);
  int dd = (int)((idx / kVTROW) % kD);
  int bh = (int)(idx / ((long)kVTROW * kD));
  int f = j / 224, p = j % 224;
  float val = 0.0f;
  if (p < kP) {
    int s = f * kP + p;
    int b = bh / kH, h = bh % kH;
    val = qkv[((size_t)(b * kN + 1 + s)) * (3 * kC) + 2 * kC + h * kD + dd];
  }
  vT[idx] = (__bf16)val;
}

// ---------------- cls token attention over all N keys; writes proj-input row b*kN
__global__ __launch_bounds__(256) void cls_attn_kernel(
    const float* __restrict__ qkv, __bf16* __restrict__ pin) {
  int bh = blockIdx.x;
  int b = bh / kH, h = bh % kH;
  int t = threadIdx.x;
  __shared__ float qs[kD];
  __shared__ float logits[kN];
  __shared__ float red[256];
  if (t < kD) qs[t] = qkv[((size_t)(b * kN)) * (3 * kC) + h * kD + t] * kScale;
  __syncthreads();
  for (int n = t; n < kN; n += 256) {
    const float* kr = qkv + ((size_t)(b * kN + n)) * (3 * kC) + kC + h * kD;
    float acc = 0.0f;
    for (int d = 0; d < kD; ++d) acc += qs[d] * kr[d];
    logits[n] = acc;
  }
  __syncthreads();
  float m = -1e30f;
  for (int n = t; n < kN; n += 256) m = fmaxf(m, logits[n]);
  red[t] = m;
  __syncthreads();
  for (int o = 128; o > 0; o >>= 1) { if (t < o) red[t] = fmaxf(red[t], red[t + o]); __syncthreads(); }
  m = red[0];
  __syncthreads();
  float s = 0.0f;
  for (int n = t; n < kN; n += 256) { float e = __expf(logits[n] - m); logits[n] = e; s += e; }
  red[t] = s;
  __syncthreads();
  for (int o = 128; o > 0; o >>= 1) { if (t < o) red[t] += red[t + o]; __syncthreads(); }
  float inv = 1.0f / red[0];
  __syncthreads();
  if (t < kD) {
    float acc = 0.0f;
    for (int n = 0; n < kN; ++n)
      acc += logits[n] * qkv[((size_t)(b * kN + n)) * (3 * kC) + 2 * kC + h * kD + t];
    pin[((size_t)(b * kN)) * kC + h * kD + t] = (__bf16)(acc * inv);
  }
}

// ---------------- fused space attention: one wave per (bh, f, 16-query tile)
// scores = q(16x64) * k^T(64x196) via WMMA; softmax; out = attn(16x224) * v(224x64)
__global__ __launch_bounds__(32) void space_attn_kernel(
    const __bf16* __restrict__ q, const __bf16* __restrict__ k,
    const __bf16* __restrict__ vT, __bf16* __restrict__ x1) {
  const int qt = blockIdx.x;   // 0..97
  const int f  = blockIdx.y;   // 0..7
  const int bh = blockIdx.z;   // 0..23
  const int lane = threadIdx.x;
  const int lr = lane & 15, lh = lane >> 4;

  __shared__ float  sc[16][224];
  __shared__ __bf16 at[16][224];

  AFrag a0, a1;   // q fragments (16 rows x K=64 -> two K32 tiles)
  {
    const __bf16* qp = q + ((size_t)bh * kS + qt * 16 + lr) * kD;
    a0.h[0] = *(const v8bf*)(qp + lh * 16);
    a0.h[1] = *(const v8bf*)(qp + lh * 16 + 8);
    a1.h[0] = *(const v8bf*)(qp + 32 + lh * 16);
    a1.h[1] = *(const v8bf*)(qp + 32 + lh * 16 + 8);
  }
  for (int jt = 0; jt < 13; ++jt) {     // 13*16 = 208 key slots (196 valid)
    v8f c = {};
    const __bf16* kp = k + ((size_t)bh * kS + f * kP + jt * 16 + lr) * kD;
    v16bf b0 = *(const v16bf*)(kp + lh * 16);
    v16bf b1 = *(const v16bf*)(kp + 32 + lh * 16);
    c = wmma_bf16(a0.v, b0, c);
    c = wmma_bf16(a1.v, b1, c);
#pragma unroll
    for (int i = 0; i < 8; ++i)
      sc[lh * 8 + i][jt * 16 + lr] = c[i] * kScale;
  }
  __syncthreads();
  if (lane < 16) {
    float mx = -1e30f;
    for (int j = 0; j < kP; ++j) mx = fmaxf(mx, sc[lane][j]);
    float s = 0.0f;
    for (int j = 0; j < kP; ++j) { float e = __expf(sc[lane][j] - mx); sc[lane][j] = e; s += e; }
    float inv = 1.0f / s;
    for (int j = 0; j < 224; ++j)
      at[lane][j] = (__bf16)(j < kP ? sc[lane][j] * inv : 0.0f);
  }
  __syncthreads();
  const int b_ = bh / kH, h_ = bh % kH;
#pragma unroll
  for (int ct = 0; ct < 4; ++ct) {      // 4 col tiles of d
    v8f c = {};
    for (int kt = 0; kt < 7; ++kt) {    // K = 224 = 7*32 (zeros past 195)
      AFrag af;
      const __bf16* ap = &at[lr][kt * 32 + lh * 16];
      af.h[0] = *(const v8bf*)(ap);
      af.h[1] = *(const v8bf*)(ap + 8);
      v16bf bf_ = *(const v16bf*)(vT + ((size_t)bh * kD + ct * 16 + lr) * kVTROW +
                                  f * 224 + kt * 32 + lh * 16);
      c = wmma_bf16(af.v, bf_, c);
    }
#pragma unroll
    for (int i = 0; i < 8; ++i) {
      int s = qt * 16 + lh * 8 + i;
      x1[(((size_t)(b_ * kS + s)) * kF + f) * kC + h_ * kD + ct * 16 + lr] = (__bf16)c[i];
    }
  }
}

// ---------------- x_diag[b, s, c] = x1[b, s, s/196, c]
__global__ __launch_bounds__(256) void gather_diag_kernel(
    const __bf16* __restrict__ x1, __bf16* __restrict__ xd) {
  long idx = (long)blockIdx.x * 256 + threadIdx.x;
  if (idx >= (long)kMS * kC) return;
  int c = (int)(idx % kC);
  int r = (int)(idx / kC);     // r = b*kS + s
  int s = r % kS;
  int f = s / kP;
  xd[idx] = x1[((size_t)r * kF + f) * kC + c];
}

// ---------------- frame attention (F=8) per (b,h,s); writes proj-input rows 1+s
__global__ __launch_bounds__(256) void attn2_kernel(
    const float* __restrict__ q2, const __bf16* __restrict__ k2,
    const __bf16* __restrict__ x1, __bf16* __restrict__ pin) {
  int idx = blockIdx.x * 256 + threadIdx.x;
  if (idx >= kB * kH * kS) return;
  int s = idx % kS;
  int h = (idx / kS) % kH;
  int b = idx / (kS * kH);
  size_t qoff = ((size_t)(b * kS + s)) * kC + h * kD;
  float logit[kF];
  float mx = -1e30f;
#pragma unroll
  for (int f = 0; f < kF; ++f) {
    size_t koff = (((size_t)(b * kS + s)) * kF + f) * kC + h * kD;
    float acc = 0.0f;
    for (int d = 0; d < kD; ++d) acc += q2[qoff + d] * (float)k2[koff + d];
    acc *= kScale;
    logit[f] = acc;
    mx = fmaxf(mx, acc);
  }
  float ssum = 0.0f;
#pragma unroll
  for (int f = 0; f < kF; ++f) { logit[f] = __expf(logit[f] - mx); ssum += logit[f]; }
  float inv = 1.0f / ssum;
  for (int d = 0; d < kD; ++d) {
    float acc = 0.0f;
#pragma unroll
    for (int f = 0; f < kF; ++f)
      acc += logit[f] * (float)x1[(((size_t)(b * kS + s)) * kF + f) * kC + h * kD + d];
    pin[((size_t)(b * kN + 1 + s)) * kC + h * kD + d] = (__bf16)(acc * inv);
  }
}

extern "C" void kernel_launch(void* const* d_in, const int* in_sizes, int n_in,
                              void* d_out, int out_size, void* d_ws, size_t ws_size,
                              hipStream_t stream) {
  (void)in_sizes; (void)n_in; (void)out_size; (void)ws_size;
  const float* x    = (const float*)d_in[0];
  const float* g1   = (const float*)d_in[1];
  const float* b1   = (const float*)d_in[2];
  const float* Wqkv = (const float*)d_in[3];
  const float* Wq   = (const float*)d_in[4];
  const float* Wkv  = (const float*)d_in[5];
  const float* Wp   = (const float*)d_in[6];
  const float* bp   = (const float*)d_in[7];
  const float* g2   = (const float*)d_in[8];
  const float* b2   = (const float*)d_in[9];
  const float* W1   = (const float*)d_in[10];
  const float* bf1  = (const float*)d_in[11];
  const float* W2   = (const float*)d_in[12];
  const float* bf2  = (const float*)d_in[13];
  float* out = (float*)d_out;

  char* ws = (char*)d_ws;
  size_t off = 0;
  auto alloc = [&](size_t bytes) -> void* {
    void* p = ws + off;
    off = (off + bytes + 255) & ~(size_t)255;
    return p;
  };
  __bf16* WqkvT = (__bf16*)alloc((size_t)(3 * kC) * kC * 2);
  __bf16* WqT   = (__bf16*)alloc((size_t)kC * kC * 2);
  __bf16* WkT   = (__bf16*)alloc((size_t)kC * kC * 2);
  __bf16* WpT   = (__bf16*)alloc((size_t)kC * kC * 2);
  __bf16* W1T   = (__bf16*)alloc((size_t)kHid * kC * 2);
  __bf16* W2T   = (__bf16*)alloc((size_t)kC * kHid * 2);
  __bf16* xn    = (__bf16*)alloc((size_t)kM1 * kC * 2);
  float*  qkv   = (float*) alloc((size_t)kM1 * (3 * kC) * 4);
  __bf16* qb    = (__bf16*)alloc((size_t)kBH * kS * kD * 2);
  __bf16* kb    = (__bf16*)alloc((size_t)(kBH * kS + 32) * kD * 2);
  __bf16* vT    = (__bf16*)alloc((size_t)kBH * kD * kVTROW * 2);
  __bf16* x1    = (__bf16*)alloc((size_t)kMX * kC * 2);
  __bf16* xd    = (__bf16*)alloc((size_t)kMS * kC * 2);
  float*  q2    = (float*) alloc((size_t)kMS * kC * 4);
  __bf16* k2    = (__bf16*)alloc((size_t)kMX * kC * 2);
  __bf16* pin   = (__bf16*)alloc((size_t)kM1 * kC * 2);
  float*  x2    = (float*) alloc((size_t)kM1 * kC * 4);
  __bf16* h2    = (__bf16*)alloc((size_t)kM1 * kC * 2);
  __bf16* gmid  = (__bf16*)alloc((size_t)kM1 * kHid * 2);

  auto gT = [](long n) { return dim3((unsigned)((n + 255) / 256)); };

  // weights -> bf16, transposed (N x K)
  transpose_w_kernel<<<gT((long)(3 * kC) * kC), 256, 0, stream>>>(Wqkv, WqkvT, kC, 3 * kC, 3 * kC, 0);
  transpose_w_kernel<<<gT((long)kC * kC), 256, 0, stream>>>(Wq, WqT, kC, kC, kC, 0);
  transpose_w_kernel<<<gT((long)kC * kC), 256, 0, stream>>>(Wkv, WkT, kC, kC, 2 * kC, 0); // only k half used
  transpose_w_kernel<<<gT((long)kC * kC), 256, 0, stream>>>(Wp, WpT, kC, kC, kC, 0);
  transpose_w_kernel<<<gT((long)kHid * kC), 256, 0, stream>>>(W1, W1T, kC, kHid, kHid, 0);
  transpose_w_kernel<<<gT((long)kC * kHid), 256, 0, stream>>>(W2, W2T, kHid, kC, kC, 0);

  // LN1 -> xn (bf16), qkv GEMM
  ln_kernel<<<dim3(kM1), 256, 0, stream>>>(x, g1, b1, xn);
  gemm_bf16_kernel<<<dim3((kM1 + 127) / 128, (3 * kC) / 64), 256, 0, stream>>>(
      xn, WqkvT, kM1, 3 * kC, kC, nullptr, nullptr, qkv, nullptr, 0);

  // gathers for attention
  gather_qk_kernel<<<gT((long)(kBH * kS + 32) * kD), 256, 0, stream>>>(qkv, qb, kb);
  gather_v_kernel<<<gT((long)kBH * kD * kVTROW), 256, 0, stream>>>(qkv, vT);

  // cls attention + fused space attention -> x1
  cls_attn_kernel<<<dim3(kBH), 256, 0, stream>>>(qkv, pin);
  space_attn_kernel<<<dim3(kS / 16, kF, kBH), 32, 0, stream>>>(qb, kb, vT, x1);

  // q2 = x_diag @ Wq ; k2 = x1 @ Wkv[:, :C]
  gather_diag_kernel<<<gT((long)kMS * kC), 256, 0, stream>>>(x1, xd);
  gemm_bf16_kernel<<<dim3((kMS + 127) / 128, kC / 64), 256, 0, stream>>>(
      xd, WqT, kMS, kC, kC, nullptr, nullptr, q2, nullptr, 0);
  gemm_bf16_kernel<<<dim3(kMX / 128, kC / 64), 256, 0, stream>>>(
      x1, WkT, kMX, kC, kC, nullptr, nullptr, nullptr, k2, 0);

  // frame attention -> proj input rows, then proj (+bp +residual x) -> x2
  attn2_kernel<<<gT((long)kB * kH * kS), 256, 0, stream>>>(q2, k2, x1, pin);
  gemm_bf16_kernel<<<dim3((kM1 + 127) / 128, kC / 64), 256, 0, stream>>>(
      pin, WpT, kM1, kC, kC, bp, x, x2, nullptr, 0);

  // MLP: LN2 -> fc1+GELU -> fc2 (+bf2 + residual x2) -> out
  ln_kernel<<<dim3(kM1), 256, 0, stream>>>(x2, g2, b2, h2);
  gemm_bf16_kernel<<<dim3((kM1 + 127) / 128, kHid / 64), 256, 0, stream>>>(
      h2, W1T, kM1, kHid, kC, bf1, nullptr, nullptr, gmid, 1);
  gemm_bf16_kernel<<<dim3((kM1 + 127) / 128, kC / 64), 256, 0, stream>>>(
      gmid, W2T, kM1, kC, kHid, bf2, x2, out, nullptr, 0);
}
